// LookupMerger_56212531970129
// MI455X (gfx1250) — matrix operand order repaired and checked by reading, not compiled, not run on hardware
//
#include <hip/hip_runtime.h>

// ---------------------------------------------------------------------------
// Problem constants (from the reference)
// ---------------------------------------------------------------------------
#define IN_DIM 2048
#define HDIM   8192
#define OUT_DIM 512
#define NBATCH 4096

typedef __attribute__((ext_vector_type(16))) __bf16 v16bf;
typedef __attribute__((ext_vector_type(8)))  __bf16 v8bf;
typedef __attribute__((ext_vector_type(4)))  __bf16 v4bf;
typedef __attribute__((ext_vector_type(8)))  float  v8f;

// GEMM tiling
#define BM 128
#define BN 128
#define BK 32
#define LDSS 40   // padded LDS row stride (elements): 80B -> conflict-free frags

// ---------------------------------------------------------------------------
// c19 activation; c already clamped, invc = 1/c precomputed per column.
// Parity of floor via cheap floor test instead of fmodf.
// ---------------------------------------------------------------------------
__device__ __forceinline__ float c19_act(float x, float c, float invc,
                                         float rho) {
    float L = 6.0f * c;
    float s = x * invc;
    float n = floorf(s);
    float t = s - n;
    float h = t * (1.0f - t);
    float nh = n * 0.5f;
    float sgn = (floorf(nh) == nh) ? 1.0f : -1.0f;
    float interior = c * (sgn * h + rho * h * h);
    return (x >= L) ? (x - L) : ((x <= -L) ? (x + L) : interior);
}

// ---------------------------------------------------------------------------
// f32 -> bf16 elementwise convert (vectorized x4)
// ---------------------------------------------------------------------------
__global__ void f32_to_bf16_kernel(const float* __restrict__ in,
                                   __bf16* __restrict__ out, long n) {
    long i = ((long)blockIdx.x * blockDim.x + threadIdx.x) * 4;
    if (i + 3 < n) {
        float4 v = *(const float4*)(in + i);
        v4bf o = { (__bf16)v.x, (__bf16)v.y, (__bf16)v.z, (__bf16)v.w };
        *(v4bf*)(out + i) = o;
    }
}

// ---------------------------------------------------------------------------
// Dequant: W = mask ? codebook[idx] : Wfloat, emitted as bf16 in BOTH
// row-major (W, [R x C]) and transposed (WT, [C x R]) layouts via an LDS
// tiled transpose. block = (32, 8); grid = (C/32, R/32).
// ---------------------------------------------------------------------------
__global__ void dequant_tr_kernel(const int* __restrict__ idx,
                                  const unsigned char* __restrict__ mask,
                                  const float* __restrict__ wf,
                                  const float* __restrict__ cb,
                                  __bf16* __restrict__ W,
                                  __bf16* __restrict__ WT,
                                  int R, int C) {
    __shared__ __bf16 tile[32][33];
    const int c0 = blockIdx.x * 32;
    const int r0 = blockIdx.y * 32;
    const int tx = threadIdx.x, ty = threadIdx.y;
#pragma unroll
    for (int j = 0; j < 4; ++j) {
        int r = ty + j * 8;
        long off = (long)(r0 + r) * C + (c0 + tx);
        float v = mask[off] ? cb[idx[off]] : wf[off];
        __bf16 b = (__bf16)v;
        W[off] = b;
        tile[r][tx] = b;
    }
    __syncthreads();
#pragma unroll
    for (int j = 0; j < 4; ++j) {
        int r = ty + j * 8;   // column index of W -> row of WT
        WT[(long)(c0 + r) * R + (r0 + tx)] = tile[tx][r];
    }
}

// ---------------------------------------------------------------------------
// Async global -> LDS staging helpers (CDNA5 GLOBAL_LOAD_ASYNC_TO_LDS,
// tracked by ASYNCcnt). ldsOff is the 32-bit LDS byte offset (low 32 bits of
// the generic shared-memory pointer); gaddr is a per-lane 64-bit address.
// Each instruction moves 16B per lane straight into LDS, no VGPR staging.
// ---------------------------------------------------------------------------
__device__ __forceinline__ void async_load_b128(unsigned ldsOff,
                                                const void* gptr) {
    asm volatile("global_load_async_to_lds_b128 %0, %1, off"
                 :
                 : "v"(ldsOff), "v"((unsigned long long)gptr)
                 : "memory");
}
__device__ __forceinline__ void wait_async0() {
    asm volatile("s_wait_asynccnt 0x0" ::: "memory");
}

// ---------------------------------------------------------------------------
// bf16 WMMA GEMM:  C[M x N] = A[M x K] * B[K x N] (+bias, +epilogue)
//   A   : row-major bf16, stride K
//   Bcm : B supplied COLUMN-major, i.e. Bcm[n*K + k] = B[k][n] (stride K)
// Epilogue template:
//   0: v = c19(acc + bias[col], ...); store bf16
//   1: v = acc + bias[col];           store f32 AND bf16
//   2: v = acc + bias[col];           store bf16
//   3: v = acc + bias[col];           store f32
// 256 threads = 8 waves (2 x 4); wave tile 64 x 32 = 4 x 2 WMMA accumulators.
// Double-buffered LDS + async loads => one barrier per K-step.
// ---------------------------------------------------------------------------
template <int EPI>
__global__ __launch_bounds__(256, 2)
void gemm_bf16_wmma(const __bf16* __restrict__ A,
                    const __bf16* __restrict__ Bcm,
                    int K, int N,
                    const float* __restrict__ bias,
                    const float* __restrict__ cParam,
                    const float* __restrict__ rhoParam,
                    float* __restrict__ outF,
                    __bf16* __restrict__ outB) {
    __shared__ __bf16 sA[2][BM * LDSS];
    __shared__ __bf16 sB[2][BN * LDSS];

    const int tid  = threadIdx.x;
    const int lane = tid & 31;
    const int wave = tid >> 5;
    const int waveM = wave >> 2;   // 0..1
    const int waveN = wave & 3;    // 0..3

    const long mBase = (long)blockIdx.y * BM;
    const long nBase = (long)blockIdx.x * BN;

    // Staging: each thread moves 2 chunks of 8 bf16 per tile (rows r, r+64)
    const int sRow = tid >> 2;          // 0..63
    const int sCol = (tid & 3) * 8;     // 0,8,16,24

    const __bf16* gA = A   + (mBase + sRow) * (long)K + sCol;
    const __bf16* gB = Bcm + (nBase + sRow) * (long)K + sCol;
    const long rowSkip = (long)64 * K;

    // Per-thread LDS byte offsets of the 4 staging chunks, per buffer.
    const unsigned chunkOff = (unsigned)((sRow * LDSS + sCol) * 2);
    const unsigned chunkOff64 = chunkOff + (unsigned)(64 * LDSS * 2);
    unsigned ldsA[2], ldsB[2];
    ldsA[0] = (unsigned)(size_t)(&sA[0][0]);
    ldsA[1] = (unsigned)(size_t)(&sA[1][0]);
    ldsB[0] = (unsigned)(size_t)(&sB[0][0]);
    ldsB[1] = (unsigned)(size_t)(&sB[1][0]);

    auto stageAsync = [&](int kb, int buf) {
        const __bf16* pa = gA + (long)kb * BK;
        const __bf16* pb = gB + (long)kb * BK;
        async_load_b128(ldsA[buf] + chunkOff,   pa);
        async_load_b128(ldsA[buf] + chunkOff64, pa + rowSkip);
        async_load_b128(ldsB[buf] + chunkOff,   pb);
        async_load_b128(ldsB[buf] + chunkOff64, pb + rowSkip);
    };

    // Fragment loads per the CDNA5 wave32 16-bit layouts.
    const int fr = lane & 15;
    const int kh = lane >> 4;
    auto loadAfrag = [&](int buf, int mt) -> v16bf {
        const __bf16* p = &sA[buf][(waveM * 64 + mt * 16 + fr) * LDSS];
        v8bf lo = *(const v8bf*)(p + kh * 8);          // K in [0..7]   (+8 if kh)
        v8bf hi = *(const v8bf*)(p + 16 + kh * 8);     // K in [16..23] (+8 if kh)
        return __builtin_shufflevector(lo, hi, 0,1,2,3,4,5,6,7,
                                               8,9,10,11,12,13,14,15);
    };
    auto loadBfrag = [&](int buf, int nt) -> v16bf {
        const __bf16* p = &sB[buf][(waveN * 32 + nt * 16 + fr) * LDSS];
        v8bf lo = *(const v8bf*)(p + kh * 8);          // K in [0..7]   (+8 if kh)
        v8bf hi = *(const v8bf*)(p + 16 + kh * 8);     // K in [16..23] (+8 if kh)
        return __builtin_shufflevector(lo, hi, 0,1,2,3,4,5,6,7,
                                               8,9,10,11,12,13,14,15);
    };

    v8f acc[4][2] = {};

    const int KB = K / BK;
    stageAsync(0, 0);
    wait_async0();
    __syncthreads();

    for (int kb = 0; kb < KB; ++kb) {
        const int cb = kb & 1;
        if (kb + 1 < KB) stageAsync(kb + 1, cb ^ 1);   // overlap with compute

        v16bf af[4];
#pragma unroll
        for (int mt = 0; mt < 4; ++mt) af[mt] = loadAfrag(cb, mt);
#pragma unroll
        for (int nt = 0; nt < 2; ++nt) {
            v16bf bfv = loadBfrag(cb, nt);
#pragma unroll
            for (int mt = 0; mt < 4; ++mt) {
                acc[mt][nt] = __builtin_amdgcn_wmma_f32_16x16x32_bf16(
                    false, af[mt], false, bfv, (short)0, acc[mt][nt],
                    false, false);
            }
        }
        wait_async0();        // next tile fully in LDS
        __syncthreads();      // single barrier per K-step
    }

    // Epilogue. C layout: VGPR g -> row = g + 8*(lane>=16); col = lane&15.
    const int colIn   = lane & 15;
    const int rowHalf = (lane >> 4) * 8;
#pragma unroll
    for (int nt = 0; nt < 2; ++nt) {
        long col = nBase + waveN * 32 + nt * 16 + colIn;
        float bv = bias[col];
        float cv = 0.0f, icv = 0.0f, rv = 0.0f;
        if (EPI == 0) {
            cv  = fmaxf(cParam[col], 0.1f);
            icv = 1.0f / cv;                 // one divide per column tile
            rv  = fmaxf(rhoParam[col], 0.0f);
        }
#pragma unroll
        for (int mt = 0; mt < 4; ++mt) {
#pragma unroll
            for (int g = 0; g < 8; ++g) {
                long row = mBase + waveM * 64 + mt * 16 + rowHalf + g;
                float v = acc[mt][nt][g] + bv;
                if (EPI == 0) v = c19_act(v, cv, icv, rv);
                if (EPI == 1 || EPI == 3) outF[row * (long)N + col] = v;
                if (EPI == 0 || EPI == 1 || EPI == 2)
                    outB[row * (long)N + col] = (__bf16)v;
            }
        }
    }
}

// ---------------------------------------------------------------------------
// Host launcher
// ---------------------------------------------------------------------------
extern "C" void kernel_launch(void* const* d_in, const int* in_sizes, int n_in,
                              void* d_out, int out_size, void* d_ws,
                              size_t ws_size, hipStream_t stream) {
    const float* x       = (const float*)d_in[0];
    const float* cb1     = (const float*)d_in[1];
    const float* cb2     = (const float*)d_in[2];
    const float* W1f     = (const float*)d_in[3];
    const float* W2f     = (const float*)d_in[4];
    const float* b1      = (const float*)d_in[5];
    const float* b2      = (const float*)d_in[6];
    const float* db1     = (const float*)d_in[7];
    const float* db2     = (const float*)d_in[8];
    const float* c_raw   = (const float*)d_in[9];
    const float* rho_raw = (const float*)d_in[10];
    const int*   W1i     = (const int*)d_in[11];
    const int*   W2i     = (const int*)d_in[12];
    const unsigned char* W1m = (const unsigned char*)d_in[13];
    const unsigned char* W2m = (const unsigned char*)d_in[14];

    // Workspace layout (bf16), 256B aligned
    char* ws = (char*)d_ws;
    size_t off = 0;
    auto wsAlloc = [&](size_t bytes) {
        char* p = ws + off;
        off += (bytes + 255) & ~(size_t)255;
        return p;
    };
    __bf16* xb   = (__bf16*)wsAlloc((size_t)NBATCH * IN_DIM * 2);  // x bf16
    __bf16* W1b  = (__bf16*)wsAlloc((size_t)IN_DIM * HDIM * 2);    // W1 row-major
    __bf16* W1tb = (__bf16*)wsAlloc((size_t)IN_DIM * HDIM * 2);    // W1^T row-major
    __bf16* W2b  = (__bf16*)wsAlloc((size_t)HDIM * OUT_DIM * 2);   // W2 row-major
    __bf16* W2tb = (__bf16*)wsAlloc((size_t)HDIM * OUT_DIM * 2);   // W2^T row-major
    __bf16* hb   = (__bf16*)wsAlloc((size_t)NBATCH * HDIM * 2);    // h, reused as u
    __bf16* zb   = (__bf16*)wsAlloc((size_t)NBATCH * OUT_DIM * 2); // z bf16
    (void)ws_size;

    float* decOut = (float*)d_out;                        // [N, IN_DIM]
    float* zOut   = decOut + (size_t)NBATCH * IN_DIM;     // [N, OUT_DIM]

    // 1) x -> bf16
    {
        long n = (long)NBATCH * IN_DIM;
        f32_to_bf16_kernel<<<(unsigned)(n / 4 / 256), 256, 0, stream>>>(x, xb, n);
    }
    // 2) dequant W1 -> W1b + W1tb ; W2 -> W2b + W2tb
    dequant_tr_kernel<<<dim3(HDIM / 32, IN_DIM / 32), dim3(32, 8), 0, stream>>>(
        W1i, W1m, W1f, cb1, W1b, W1tb, IN_DIM, HDIM);
    dequant_tr_kernel<<<dim3(OUT_DIM / 32, HDIM / 32), dim3(32, 8), 0, stream>>>(
        W2i, W2m, W2f, cb2, W2b, W2tb, HDIM, OUT_DIM);

    // 3) h = c19(x @ W1 + b1)          [N x H], B col-major = W1^T row-major
    gemm_bf16_wmma<0><<<dim3(HDIM / BN, NBATCH / BM), 256, 0, stream>>>(
        xb, W1tb, IN_DIM, HDIM, b1, c_raw, rho_raw, nullptr, hb);

    // 4) z = h @ W2 + b2               [N x OUT], B col-major = W2^T row-major
    gemm_bf16_wmma<1><<<dim3(OUT_DIM / BN, NBATCH / BM), 256, 0, stream>>>(
        hb, W2tb, HDIM, OUT_DIM, b2, nullptr, nullptr, zOut, zb);

    // 5) u = z @ W2^T + db1            [N x H], B = W2^T, col-major = W2 row-major
    gemm_bf16_wmma<2><<<dim3(HDIM / BN, NBATCH / BM), 256, 0, stream>>>(
        zb, W2b, OUT_DIM, HDIM, db1, nullptr, nullptr, nullptr, hb);

    // 6) dec = u @ W1^T + db2          [N x IN], B = W1^T, col-major = W1 row-major
    gemm_bf16_wmma<3><<<dim3(IN_DIM / BN, NBATCH / BM), 256, 0, stream>>>(
        hb, W1b, HDIM, IN_DIM, db2, nullptr, nullptr, decOut, nullptr);
}
